// BertSelfAttention_25237227831508
// MI455X (gfx1250) — matrix operand order repaired and checked by reading, not compiled, not run on hardware
//
#include <hip/hip_runtime.h>

typedef __attribute__((ext_vector_type(16))) __bf16 v16bf;
typedef __attribute__((ext_vector_type(2)))  __bf16 v2bf;
typedef __attribute__((ext_vector_type(8)))  float  v8f;
typedef __attribute__((ext_vector_type(4)))  int    v4i;
typedef __attribute__((address_space(1))) v4i* as1_v4p;
typedef __attribute__((address_space(3))) v4i* as3_v4p;

// xor-shuffle via ds_swizzle (group-of-32: and=0x1f, or=0, xor=m). Mask <= 8
// keeps the shuffle inside a 16-lane half (bit4 untouched).
#define SWZ_XOR_F(v, m) \
  __int_as_float(__builtin_amdgcn_ds_swizzle(__float_as_int(v), (((m) << 10) | 0x1f)))

static constexpr int HID    = 1024;
static constexpr int BATCH  = 8;
static constexpr int SEQ    = 1024;
static constexpr int NHEAD  = 16;
static constexpr int DH     = 64;
static constexpr int BS     = BATCH * SEQ;   // 8192 rows

// ---------------- CDNA5 async global->LDS staging (ASYNCcnt), with fallback
#if defined(__has_builtin)
#if __has_builtin(__builtin_amdgcn_global_load_async_to_lds_b128)
#define HAVE_ASYNC_LDS 1
#endif
#endif
#ifndef HAVE_ASYNC_LDS
#define HAVE_ASYNC_LDS 0
#endif

__device__ __forceinline__ void stage16(const __bf16* g, __bf16* l) {
#if HAVE_ASYNC_LDS
  // Param 0 is AS1 (global) int4*, param 1 is AS3 (LDS) int4*; go through
  // uintptr_t so const-ness and address-space casts are reinterprets.
  __builtin_amdgcn_global_load_async_to_lds_b128(
      (as1_v4p)(uintptr_t)g, (as3_v4p)(unsigned int)(uintptr_t)l, 0, 0);
#else
  *(uint4*)l = *(const uint4*)g;
#endif
}

__device__ __forceinline__ void stage_wait() {
#if HAVE_ASYNC_LDS
#if defined(__has_builtin) && __has_builtin(__builtin_amdgcn_s_wait_asynccnt)
  __builtin_amdgcn_s_wait_asynccnt(0);
#else
  asm volatile("s_wait_asynccnt 0x0" ::: "memory");
#endif
#endif
}

// ---------------------------------------------------------------- fp32->bf16
__global__ void cvt_bf16_kernel(const float* __restrict__ in,
                                __bf16* __restrict__ out, int n) {
  int i = blockIdx.x * blockDim.x + threadIdx.x;
  if (i < n) out[i] = (__bf16)in[i];
}

// --------------------------------------------------- WMMA fragment loaders
// A-matrix 16x32 bf16: lane<16 -> M=lane, K=0..15 (2/VGPR),
// lanes 16..31 K=16..31.
__device__ __forceinline__ v16bf load_a_bf16(const __bf16* base, int ld) {
  int lane = threadIdx.x & 31;
  int hf   = lane >> 4;
  int m    = lane & 15;
  v16bf a;
#pragma unroll
  for (int j = 0; j < 8; ++j) {
    int k   = 2 * j + hf * 16;
    v2bf p  = *(const v2bf*)(base + m * ld + k);
    a[2 * j]     = p.x;
    a[2 * j + 1] = p.y;
  }
  return a;
}

// B-matrix 32x16 bf16, source memory holds rows n with k contiguous:
// B[k][n] = base[n*ld + k]; lane<16 -> n=lane, K=0..15 (2/VGPR), lanes 16..31 K=16..31.
__device__ __forceinline__ v16bf load_b_nk(const __bf16* base, int ld) {
  int lane = threadIdx.x & 31;
  int hf   = lane >> 4;
  int n    = lane & 15;
  v16bf b;
#pragma unroll
  for (int j = 0; j < 8; ++j) {
    int k  = 2 * j + hf * 16;
    v2bf p = *(const v2bf*)(base + n * ld + k);
    b[2 * j]     = p.x;
    b[2 * j + 1] = p.y;
  }
  return b;
}

// B-matrix 32x16 bf16, source memory laid out [k][n]: B[k][n] = base[k*ld + n].
__device__ __forceinline__ v16bf load_b_kn(const __bf16* base, int ld) {
  int lane = threadIdx.x & 31;
  int hf   = lane >> 4;
  int n    = lane & 15;
  v16bf b;
#pragma unroll
  for (int j = 0; j < 8; ++j) {
    int k = 2 * j + hf * 16;
    b[2 * j]     = base[k * ld + n];
    b[2 * j + 1] = base[(k + 1) * ld + n];
  }
  return b;
}

__device__ __forceinline__ v8f wmma_bf16(v16bf a, v16bf b, v8f c) {
  return __builtin_amdgcn_wmma_f32_16x16x32_bf16(
      false, a, false, b, (short)0, c, false, false);
}

// ------------------------------------------------------------- QKV GEMM
// y = x @ W^T + bias ; out stored bf16 in [B, H, S, D] per-head layout.
// Block: 256 thr (8 waves), tile 256(M) x 64(N), K step 32, LDS double-buffered.
// Each wave: 32x64 strip = 2 A-frags x 4 B-frags = 8 WMMAs / K-step.
__global__ __launch_bounds__(256) void qkv_gemm_kernel(
    const __bf16* __restrict__ xb,
    const __bf16* __restrict__ wq, const __bf16* __restrict__ wk,
    const __bf16* __restrict__ wv,
    const float* __restrict__ bq, const float* __restrict__ bk,
    const float* __restrict__ bv,
    __bf16* __restrict__ qo, __bf16* __restrict__ ko, __bf16* __restrict__ vo) {
  const int z = blockIdx.z;
  const __bf16* W    = (z == 0) ? wq : (z == 1) ? wk : wv;
  const float*  bias = (z == 0) ? bq : (z == 1) ? bk : bv;
  __bf16*       out  = (z == 0) ? qo : (z == 1) ? ko : vo;

  __shared__ __align__(16) __bf16 xs[2][256 * 32];   // 2 x 16 KB
  __shared__ __align__(16) __bf16 wsh[2][64 * 32];   // 2 x  4 KB

  const int tid   = threadIdx.x;
  const int wv_id = tid >> 5;       // wave 0..7
  const int lane  = tid & 31;
  const int m_blk = blockIdx.y * 256;
  const int n_blk = blockIdx.x * 64;

  // Per-thread staging slots.
  const int xr[4] = {(tid + 0) >> 2, (tid + 256) >> 2, (tid + 512) >> 2,
                     (tid + 768) >> 2};
  const int xc = (tid & 3) * 8;
  const int wr = tid >> 2, wc = (tid & 3) * 8;

  v8f acc[2][4] = {};

  // issue tile kk=0 into buffer 0
#pragma unroll
  for (int i = 0; i < 4; ++i)
    stage16(xb + (size_t)(m_blk + xr[i]) * HID + xc, &xs[0][xr[i] * 32 + xc]);
  stage16(W + (size_t)(n_blk + wr) * HID + wc, &wsh[0][wr * 32 + wc]);

  for (int kk = 0; kk < HID; kk += 32) {
    const int p = (kk >> 5) & 1;
    stage_wait();            // this wave's async copies for tile kk landed
    __syncthreads();         // everyone's copies landed; prev buf consumed
    if (kk + 32 < HID) {     // issue next tile into the other buffer
      const int kn = kk + 32;
#pragma unroll
      for (int i = 0; i < 4; ++i)
        stage16(xb + (size_t)(m_blk + xr[i]) * HID + kn + xc,
                &xs[p ^ 1][xr[i] * 32 + xc]);
      stage16(W + (size_t)(n_blk + wr) * HID + kn + wc,
              &wsh[p ^ 1][wr * 32 + wc]);
    }
    // compute from buffer p while next tile is in flight
    v16bf a0 = load_a_bf16(&xs[p][(wv_id * 32 + 0) * 32], 32);
    v16bf a1 = load_a_bf16(&xs[p][(wv_id * 32 + 16) * 32], 32);
#pragma unroll
    for (int t = 0; t < 4; ++t) {
      v16bf b = load_b_nk(&wsh[p][t * 16 * 32], 32);
      acc[0][t] = wmma_bf16(a0, b, acc[0][t]);
      acc[1][t] = wmma_bf16(a1, b, acc[1][t]);
    }
  }

  // epilogue: bias + bf16 store into [B,H,S,D]
  const int hf = lane >> 4, nl = lane & 15;
#pragma unroll
  for (int af = 0; af < 2; ++af) {
#pragma unroll
    for (int t = 0; t < 4; ++t) {
      int o = n_blk + t * 16 + nl;
      float bsv = bias[o];
      int h = o >> 6, d = o & 63;
#pragma unroll
      for (int i = 0; i < 8; ++i) {
        int row = m_blk + wv_id * 32 + af * 16 + hf * 8 + i;
        int bb = row >> 10, s = row & 1023;
        float y = acc[af][t][i] + bsv;
        out[((size_t)(bb * NHEAD + h) * SEQ + s) * DH + d] = (__bf16)y;
      }
    }
  }
}

// --------------------------------------------------------- flash attention
// grid (S/128, B*H); 256 thr / 8 waves; wave owns 16 query rows.
// K/V staged 32 keys at a time, LDS double-buffered via async copies.
__global__ __launch_bounds__(256) void attn_kernel(
    const __bf16* __restrict__ qb, const __bf16* __restrict__ kb,
    const __bf16* __restrict__ vb, const float* __restrict__ mask,
    float* __restrict__ out) {
  const int bh = blockIdx.y;
  const int b  = bh >> 4, h = bh & 15;
  const int tid = threadIdx.x, wv_id = tid >> 5, lane = tid & 31;
  const int hf = lane >> 4, nl = lane & 15;

  const __bf16* qbase = qb + (size_t)bh * SEQ * DH;
  const __bf16* kbase = kb + (size_t)bh * SEQ * DH;
  const __bf16* vbase = vb + (size_t)bh * SEQ * DH;
  const int q0 = blockIdx.x * 128 + wv_id * 16;

  __shared__ __align__(16) __bf16 ksm[2][32 * 64];   // 2 x 4 KB
  __shared__ __align__(16) __bf16 vsm[2][32 * 64];   // 2 x 4 KB
  __shared__ __align__(16) __bf16 psm[8 * 16 * 32];  // per-wave P scratch
  __bf16* myps = psm + wv_id * 512;

  const int sr = tid >> 3, sc = (tid & 7) * 8;  // staging slot (1 b128 each)

  // Q fragments (16 rows x 64 d, as two K=32 A-frags), loaded once
  v16bf aq0 = load_a_bf16(qbase + (size_t)q0 * DH + 0, DH);
  v16bf aq1 = load_a_bf16(qbase + (size_t)q0 * DH + 32, DH);

  v8f o0 = {}, o1 = {}, o2 = {}, o3 = {};
  float mrow[8], lrow[8];
#pragma unroll
  for (int i = 0; i < 8; ++i) { mrow[i] = -1e30f; lrow[i] = 0.f; }

  // issue first K/V tile into buffer 0
  stage16(kbase + (size_t)sr * DH + sc, &ksm[0][sr * DH + sc]);
  stage16(vbase + (size_t)sr * DH + sc, &vsm[0][sr * DH + sc]);

  for (int j0 = 0; j0 < SEQ; j0 += 32) {
    const int p = (j0 >> 5) & 1;
    stage_wait();
    __syncthreads();
    if (j0 + 32 < SEQ) {
      stage16(kbase + (size_t)(j0 + 32 + sr) * DH + sc, &ksm[p ^ 1][sr * DH + sc]);
      stage16(vbase + (size_t)(j0 + 32 + sr) * DH + sc, &vsm[p ^ 1][sr * DH + sc]);
    }

    // scores for 32 keys: two 16x16 C tiles, each accumulated over d in 2 WMMAs
    v8f s0 = {}, s1 = {};
    s0 = wmma_bf16(aq0, load_b_nk(&ksm[p][0 * DH + 0],  DH), s0);
    s0 = wmma_bf16(aq1, load_b_nk(&ksm[p][0 * DH + 32], DH), s0);
    s1 = wmma_bf16(aq0, load_b_nk(&ksm[p][16 * DH + 0],  DH), s1);
    s1 = wmma_bf16(aq1, load_b_nk(&ksm[p][16 * DH + 32], DH), s1);

    float mk0 = mask[b * SEQ + j0 + nl];
    float mk1 = mask[b * SEQ + j0 + 16 + nl];

    // online softmax per C-row (row = hf*8 + i, columns spread over 16 lanes)
#pragma unroll
    for (int i = 0; i < 8; ++i) {
      float v0 = s0[i] * 0.125f + mk0;   // 1/sqrt(64)
      float v1 = s1[i] * 0.125f + mk1;
      float tmax = fmaxf(v0, v1);
      tmax = fmaxf(tmax, SWZ_XOR_F(tmax, 1));
      tmax = fmaxf(tmax, SWZ_XOR_F(tmax, 2));
      tmax = fmaxf(tmax, SWZ_XOR_F(tmax, 4));
      tmax = fmaxf(tmax, SWZ_XOR_F(tmax, 8));
      float nm = fmaxf(mrow[i], tmax);
      float p0 = __expf(v0 - nm);
      float p1 = __expf(v1 - nm);
      float rs = p0 + p1;
      rs += SWZ_XOR_F(rs, 1);
      rs += SWZ_XOR_F(rs, 2);
      rs += SWZ_XOR_F(rs, 4);
      rs += SWZ_XOR_F(rs, 8);
      float sc2 = __expf(mrow[i] - nm);
      lrow[i] = lrow[i] * sc2 + rs;
      mrow[i] = nm;
      o0[i] *= sc2; o1[i] *= sc2; o2[i] *= sc2; o3[i] *= sc2;
      int r = hf * 8 + i;  // re-shape P from C-layout to A-layout via LDS
      myps[r * 32 + nl]      = (__bf16)p0;
      myps[r * 32 + 16 + nl] = (__bf16)p1;
    }

    // O += P (16x32) x V (32x64)
    v16bf pa = load_a_bf16(myps, 32);
    o0 = wmma_bf16(pa, load_b_kn(&vsm[p][0],  DH), o0);
    o1 = wmma_bf16(pa, load_b_kn(&vsm[p][16], DH), o1);
    o2 = wmma_bf16(pa, load_b_kn(&vsm[p][32], DH), o2);
    o3 = wmma_bf16(pa, load_b_kn(&vsm[p][48], DH), o3);
  }

  // normalize and store fp32 [B, S, H*D]
#pragma unroll
  for (int i = 0; i < 8; ++i) {
    float inv = 1.0f / lrow[i];
    int s = q0 + hf * 8 + i;
    size_t base_o = ((size_t)(b * SEQ + s)) * HID + h * DH + nl;
    out[base_o + 0]  = o0[i] * inv;
    out[base_o + 16] = o1[i] * inv;
    out[base_o + 32] = o2[i] * inv;
    out[base_o + 48] = o3[i] * inv;
  }
}

// ------------------------------------------------------------------ launch
extern "C" void kernel_launch(void* const* d_in, const int* in_sizes, int n_in,
                              void* d_out, int out_size, void* d_ws, size_t ws_size,
                              hipStream_t stream) {
  const float* x    = (const float*)d_in[0];
  const float* mask = (const float*)d_in[1];
  const float* Wq   = (const float*)d_in[2];
  const float* bq   = (const float*)d_in[3];
  const float* Wk   = (const float*)d_in[4];
  const float* bk   = (const float*)d_in[5];
  const float* Wv   = (const float*)d_in[6];
  const float* bv   = (const float*)d_in[7];

  char* wsp = (char*)d_ws;
  const size_t XB_BYTES  = (size_t)BS * HID * 2;                   // 16 MB
  const size_t W_BYTES   = (size_t)HID * HID * 2;                  //  2 MB
  const size_t QKV_BYTES = (size_t)BATCH * NHEAD * SEQ * DH * 2;   // 16 MB
  __bf16* xb  = (__bf16*)(wsp);
  __bf16* wqb = (__bf16*)(wsp + XB_BYTES);
  __bf16* wkb = (__bf16*)(wsp + XB_BYTES + W_BYTES);
  __bf16* wvb = (__bf16*)(wsp + XB_BYTES + 2 * W_BYTES);
  __bf16* qbf = (__bf16*)(wsp + XB_BYTES + 3 * W_BYTES);
  __bf16* kbf = (__bf16*)(wsp + XB_BYTES + 3 * W_BYTES + QKV_BYTES);
  __bf16* vbf = (__bf16*)(wsp + XB_BYTES + 3 * W_BYTES + 2 * QKV_BYTES);

  const int nx = BS * HID;       // 8388608
  const int nw = HID * HID;      // 1048576
  cvt_bf16_kernel<<<(nx + 255) / 256, 256, 0, stream>>>(x, xb, nx);
  cvt_bf16_kernel<<<(nw + 255) / 256, 256, 0, stream>>>(Wq, wqb, nw);
  cvt_bf16_kernel<<<(nw + 255) / 256, 256, 0, stream>>>(Wk, wkb, nw);
  cvt_bf16_kernel<<<(nw + 255) / 256, 256, 0, stream>>>(Wv, wvb, nw);

  qkv_gemm_kernel<<<dim3(HID / 64, BS / 256, 3), 256, 0, stream>>>(
      xb, wqb, wkb, wvb, bq, bk, bv, qbf, kbf, vbf);

  attn_kernel<<<dim3(SEQ / 128, BATCH * NHEAD), 256, 0, stream>>>(
      qbf, kbf, vbf, mask, (float*)d_out);
}
